// SwinOccFormerBlock_16844861735227
// MI455X (gfx1250) — matrix operand order repaired and checked by reading, not compiled, not run on hardware
//
#include <hip/hip_runtime.h>
#include <hip/hip_bf16.h>
#include <math.h>
#include <stdint.h>

// SwinOccFormer block for MI455X (gfx1250, wave32, WMMA).
// Large GEMMs: v_wmma_f32_16x16x32_bf16, 128x128x32 block tiles, 8 waves,
// 8 WMMA per wave per K-step, double-buffered LDS, async A-tile staging via
// global_load_async_to_lds_b128 (+ s_wait_asynccnt).
// Workspace requirement: ~310 MB (lifetime-reused regions).

#define USE_ASYNC_LDS 1

#define DEVFN __device__ __forceinline__

typedef __attribute__((ext_vector_type(16))) __bf16 v16bf;
typedef __attribute__((ext_vector_type(8)))  float  v8f;

union FragAB { v16bf v; unsigned int u[8]; };

constexpr int Bn   = 8;
constexpr int Cc   = 512;
constexpr int Hh   = 56;
constexpr int Wimg = 56;
constexpr int NHh  = 16;
constexpr int Ll   = 64;
constexpr int HDd  = 32;
constexpr int Tt   = 49;
constexpr int NWw  = 64;            // (56/7)^2 windows per image
constexpr int BWw  = Bn * NWw;      // 512 windows
constexpr int NTOK = BWw * Tt;      // 25088 token rows
constexpr int NPIX = Hh * Wimg;     // 3136

DEVFN v8f wmma_bf16(const v16bf& a, const v16bf& b, v8f c) {
  return __builtin_amdgcn_wmma_f32_16x16x32_bf16(false, a, false, b, (short)0, c,
                                                 false, false);
}

DEVFN float gelu_exact(float x) {
  return 0.5f * x * (1.0f + erff(x * 0.70710678118654752f));
}

// ---------------------------------------------------------------------------
// Tiled WMMA GEMM: C[M,N] = A[M,K](bf16) @ W[K,N](fp32, row stride ldw) + bias
// BM=BN=128, BK=32. 256 threads = 8 waves in a 4x2 grid; each wave owns a
// 32x64 output sub-tile (2x4 WMMA tiles -> 8 WMMA per K-step).
// Double-buffered LDS: one barrier per K-step; A staged by async-to-LDS.
// EPI: 0 = bias -> bf16 out, 1 = bias+GELU -> bf16 out, 2 = bias -> fp32 out.
// Requires M%128==0, N%128==0, K%32==0 (true for every GEMM here).
// ---------------------------------------------------------------------------
constexpr int BM = 128, BN = 128, BK = 32;
constexpr int LDA = BK + 8;   // 40 bf16 = 80B row stride (keeps 16B alignment)
constexpr int LDB = BK + 8;

template <int EPI>
__global__ __launch_bounds__(256) void k_gemm(const __bf16* __restrict__ A,
                                              const float* __restrict__ W,
                                              int ldw,
                                              const float* __restrict__ bias,
                                              void* __restrict__ outp,
                                              int ldout, int K) {
  __shared__ __bf16 sA[2][BM * LDA];      // row-major  [m][k]
  __shared__ __bf16 sB[2][BN * LDB];      // transposed [n][k]

  const int tid   = threadIdx.x;
  const int lane  = tid & 31;
  const int wave  = tid >> 5;
  const int tileM = blockIdx.y * BM;
  const int tileN = blockIdx.x * BN;
  const int wm    = (wave >> 1) * 32;     // wave row block: 0,32,64,96
  const int wn    = (wave & 1) * 64;      // wave col block: 0,64
  const int hl    = lane >> 4;            // lane half (ISA 16-bit layouts)
  const int r     = lane & 15;

  v8f acc[2][4];
#pragma unroll
  for (int mi = 0; mi < 2; ++mi)
#pragma unroll
    for (int ni = 0; ni < 4; ++ni) acc[mi][ni] = (v8f){};

  // A-tile staging: thread -> (row, 16-elem chunk); 2 x b128 per thread.
  const int ar = tid >> 1;                // 0..127
  const int ac = (tid & 1) * 16;          // 0 or 16
  // B-tile staging: thread -> (k row, 16 cols); 4 x float4 per thread.
  const int bk = tid >> 3;                // 0..31
  const int bn0 = (tid & 7) * 16;         // 0..112

  const int nk = K / BK;

  // ---- stage tile 0 into buffer 0 -------------------------------------
  {
    const __bf16* ga = A + (size_t)(tileM + ar) * K + ac;
#if USE_ASYNC_LDS
    unsigned lo = (unsigned)(uintptr_t)&sA[0][ar * LDA + ac];
    asm volatile("global_load_async_to_lds_b128 %0, %1, off"
                 :: "v"(lo), "v"(ga) : "memory");
    asm volatile("global_load_async_to_lds_b128 %0, %1, off offset:16"
                 :: "v"(lo), "v"(ga) : "memory");
#else
    *(uint4*)&sA[0][ar * LDA + ac]     = *(const uint4*)ga;
    *(uint4*)&sA[0][ar * LDA + ac + 8] = *(const uint4*)(ga + 8);
#endif
    const float* gw = W + (size_t)bk * ldw + tileN + bn0;
    float4 w0 = *(const float4*)gw;
    float4 w1 = *(const float4*)(gw + 4);
    float4 w2 = *(const float4*)(gw + 8);
    float4 w3 = *(const float4*)(gw + 12);
    const float* wv = (const float*)&w0;   // w0..w3 contiguous on stack? no —
    // store explicitly instead:
    float wf[16] = {w0.x, w0.y, w0.z, w0.w, w1.x, w1.y, w1.z, w1.w,
                    w2.x, w2.y, w2.z, w2.w, w3.x, w3.y, w3.z, w3.w};
    (void)wv;
#pragma unroll
    for (int j = 0; j < 16; ++j) sB[0][(bn0 + j) * LDB + bk] = (__bf16)wf[j];
  }
#if USE_ASYNC_LDS
  asm volatile("s_wait_asynccnt 0x0" ::: "memory");
#endif
  __syncthreads();

  for (int kt = 0; kt < nk; ++kt) {
    const int cur = kt & 1, nxt = cur ^ 1;
    const bool more = (kt + 1 < nk);
    float wf[16];
    if (more) {
      const int k0 = (kt + 1) * BK;
      const __bf16* ga = A + (size_t)(tileM + ar) * K + k0 + ac;
#if USE_ASYNC_LDS
      unsigned lo = (unsigned)(uintptr_t)&sA[nxt][ar * LDA + ac];
      asm volatile("global_load_async_to_lds_b128 %0, %1, off"
                   :: "v"(lo), "v"(ga) : "memory");
      asm volatile("global_load_async_to_lds_b128 %0, %1, off offset:16"
                   :: "v"(lo), "v"(ga) : "memory");
#else
      *(uint4*)&sA[nxt][ar * LDA + ac]     = *(const uint4*)ga;
      *(uint4*)&sA[nxt][ar * LDA + ac + 8] = *(const uint4*)(ga + 8);
#endif
      const float* gw = W + (size_t)(k0 + bk) * ldw + tileN + bn0;
      float4 w0 = *(const float4*)gw;
      float4 w1 = *(const float4*)(gw + 4);
      float4 w2 = *(const float4*)(gw + 8);
      float4 w3 = *(const float4*)(gw + 12);
      wf[0] = w0.x;  wf[1] = w0.y;  wf[2] = w0.z;  wf[3] = w0.w;
      wf[4] = w1.x;  wf[5] = w1.y;  wf[6] = w1.z;  wf[7] = w1.w;
      wf[8] = w2.x;  wf[9] = w2.y;  wf[10] = w2.z; wf[11] = w2.w;
      wf[12] = w3.x; wf[13] = w3.y; wf[14] = w3.z; wf[15] = w3.w;
    }

    // ---- compute on current buffer (8 WMMA) ---------------------------
    FragAB fb[4];
#pragma unroll
    for (int ni = 0; ni < 4; ++ni) {
      const __bf16* bp = &sB[cur][(wn + ni * 16 + r) * LDB + hl * 16];
#pragma unroll
      for (int j = 0; j < 8; ++j)
        fb[ni].u[j] = *(const unsigned int*)(bp + 2 * j);
    }
#pragma unroll
    for (int mi = 0; mi < 2; ++mi) {
      FragAB fa;
      const __bf16* ap = &sA[cur][(wm + mi * 16 + r) * LDA + hl * 16];
#pragma unroll
      for (int j = 0; j < 4; ++j) {
        fa.u[j]     = *(const unsigned int*)(ap + 2 * j);
        fa.u[4 + j] = *(const unsigned int*)(ap + 8 + 2 * j);
      }
#pragma unroll
      for (int ni = 0; ni < 4; ++ni)
        acc[mi][ni] = wmma_bf16(fa.v, fb[ni].v, acc[mi][ni]);
    }

    if (more) {
#pragma unroll
      for (int j = 0; j < 16; ++j)
        sB[nxt][(bn0 + j) * LDB + bk] = (__bf16)wf[j];
#if USE_ASYNC_LDS
      asm volatile("s_wait_asynccnt 0x0" ::: "memory");
#endif
      __syncthreads();
    }
  }

  // ---- epilogue --------------------------------------------------------
#pragma unroll
  for (int mi = 0; mi < 2; ++mi) {
#pragma unroll
    for (int ni = 0; ni < 4; ++ni) {
      const int col = tileN + wn + ni * 16 + r;
      const float bc = bias[col];
#pragma unroll
      for (int j = 0; j < 8; ++j) {
        int row = tileM + wm + mi * 16 + hl * 8 + j;
        float x = acc[mi][ni][j] + bc;
        if (EPI == 1) x = gelu_exact(x);
        if (EPI == 2) {
          ((float*)outp)[(size_t)row * ldout + col] = x;
        } else {
          ((__bf16*)outp)[(size_t)row * ldout + col] = (__bf16)x;
        }
      }
    }
  }
}

// ---------------------------------------------------------------------------
// fp32 -> bf16 flat convert (for the `vector` input feeding the first GEMM)
// ---------------------------------------------------------------------------
__global__ void k_f32_to_bf16(const float* __restrict__ s, __bf16* __restrict__ d,
                              int n) {
  int i = blockIdx.x * blockDim.x + threadIdx.x;
  if (i < n) d[i] = (__bf16)s[i];
}

// ---------------------------------------------------------------------------
// Token<->pixel mapping for roll(-3,-3) + unfold (and its inverse fold+roll).
// ---------------------------------------------------------------------------
DEVFN void token_to_pixel(int rrow, int& b, int& h, int& w) {
  int bw = rrow / Tt, t = rrow % Tt;
  b = bw >> 6;
  int wh = (bw & 63) >> 3, ww = bw & 7;
  h = (wh * 7 + t / 7 + 3) % Hh;
  w = (ww * 7 + t % 7 + 3) % Wimg;
}

// tokens[row, c] = src[b, c, h, w]  (bf16), row = window-token index
__global__ __launch_bounds__(256) void k_gather_tokens(
    const float* __restrict__ src, __bf16* __restrict__ dst) {
  int rrow = blockIdx.x;
  int b, h, w;
  token_to_pixel(rrow, b, h, w);
  size_t base = ((size_t)b * Cc) * NPIX + (size_t)h * Wimg + w;
  int c0 = threadIdx.x, c1 = c0 + 256;
  dst[(size_t)rrow * Cc + c0] = (__bf16)src[base + (size_t)c0 * NPIX];
  dst[(size_t)rrow * Cc + c1] = (__bf16)src[base + (size_t)c1 * NPIX];
}

// ---------------------------------------------------------------------------
// Meta-MLP relative-position bias: biasT[h, q*49+k]  (REL_LOG built on the fly)
// ---------------------------------------------------------------------------
__global__ __launch_bounds__(256) void k_bias_mlp(const float* __restrict__ w1,
                                                  const float* __restrict__ b1,
                                                  const float* __restrict__ w2,
                                                  const float* __restrict__ b2,
                                                  float* __restrict__ biasT) {
  __shared__ float hid[256];
  int p = blockIdx.x;                 // 0..2400 (q*49+k)
  int q = p / Tt, k = p % Tt;
  float dh = (float)(q / 7 - k / 7);
  float dw = (float)(q % 7 - k % 7);
  float r0 = ((dh > 0.f) - (dh < 0.f)) * log1pf(fabsf(dh));
  float r1 = ((dw > 0.f) - (dw < 0.f)) * log1pf(fabsf(dw));
  int c = threadIdx.x;
  float hv = r0 * w1[c] + r1 * w1[256 + c] + b1[c];
  hid[c] = hv > 0.f ? hv : 0.f;
  __syncthreads();
  if (c < NHh) {
    float s = b2[c];
    for (int i = 0; i < 256; ++i) s += hid[i] * w2[i * NHh + c];
    biasT[c * (Tt * Tt) + p] = s;
  }
}

// ---------------------------------------------------------------------------
// Window attention (cosine sim, tau, bias, shift mask, softmax, attn@V).
// One block per window, looped over 16 heads. Small 49x49 einsums on VALU.
// ---------------------------------------------------------------------------
DEVFN int region_id(int ph, int pw) {
  int rh = ph < 49 ? 0 : (ph < 53 ? 1 : 2);
  int rw = pw < 49 ? 0 : (pw < 53 ? 1 : 2);
  return rh * 3 + rw;
}

__global__ __launch_bounds__(256) void k_win_attn(
    const __bf16* __restrict__ qkv, const float* __restrict__ biasT,
    const float* __restrict__ tau, __bf16* __restrict__ wout) {
  __shared__ float sq[Tt * HDd], sk[Tt * HDd], sv[Tt * HDd];
  __shared__ float ss[Tt * Tt];
  __shared__ float nq[Tt], nk[Tt];

  int bw = blockIdx.x, tid = threadIdx.x;
  int wh = (bw & 63) >> 3, ww = bw & 7;

  for (int h = 0; h < NHh; ++h) {
    float tmax = tau[h];
    tmax = tmax > 0.01f ? tmax : 0.01f;

    for (int idx = tid; idx < Tt * HDd; idx += 256) {
      int t = idx >> 5, d = idx & 31;
      size_t base = ((size_t)bw * Tt + t) * (3 * Cc) + h * HDd + d;
      sq[idx] = (float)qkv[base];
      sk[idx] = (float)qkv[base + Cc];
      sv[idx] = (float)qkv[base + 2 * Cc];
    }
    __syncthreads();

    if (tid < Tt) {
      float a = 0.f, b = 0.f;
      for (int d = 0; d < HDd; ++d) {
        float x = sq[tid * HDd + d]; a += x * x;
        float y = sk[tid * HDd + d]; b += y * y;
      }
      nq[tid] = sqrtf(a); nk[tid] = sqrtf(b);
    }
    __syncthreads();

    for (int p = tid; p < Tt * Tt; p += 256) {
      int i = p / Tt, j = p % Tt;
      float dot = 0.f;
      for (int d = 0; d < HDd; ++d) dot += sq[i * HDd + d] * sk[j * HDd + d];
      float den = nq[i] * nk[j];
      den = den > 1e-6f ? den : 1e-6f;
      float val = dot / den / tmax + biasT[h * (Tt * Tt) + p];
      int ri = region_id(wh * 7 + i / 7, ww * 7 + i % 7);
      int rj = region_id(wh * 7 + j / 7, ww * 7 + j % 7);
      if (ri != rj) val -= 100.0f;
      ss[p] = val;
    }
    __syncthreads();

    if (tid < Tt) {
      float m = -1e30f;
      for (int j = 0; j < Tt; ++j) m = fmaxf(m, ss[tid * Tt + j]);
      float s = 0.f;
      for (int j = 0; j < Tt; ++j) {
        float e = __expf(ss[tid * Tt + j] - m);
        ss[tid * Tt + j] = e; s += e;
      }
      float inv = 1.0f / s;
      for (int j = 0; j < Tt; ++j) ss[tid * Tt + j] *= inv;
    }
    __syncthreads();

    for (int idx = tid; idx < Tt * HDd; idx += 256) {
      int t = idx >> 5, d = idx & 31;
      float o = 0.f;
      for (int j = 0; j < Tt; ++j) o += ss[t * Tt + j] * sv[j * HDd + d];
      wout[((size_t)bw * Tt + t) * Cc + h * HDd + d] = (__bf16)o;
    }
    __syncthreads();
  }
}

// ---------------------------------------------------------------------------
// Cross attention vs 64 vector tokens (k2/v2 precomputed per batch).
// q2 rows already contain bias; HD^-0.5 folded into the dot here.
// ---------------------------------------------------------------------------
__global__ __launch_bounds__(256) void k_cross_attn(
    const __bf16* __restrict__ q2, const __bf16* __restrict__ k2,
    const __bf16* __restrict__ v2, __bf16* __restrict__ co) {
  __shared__ float sq[Tt * HDd];
  __shared__ float sk[Ll * HDd], sv[Ll * HDd];
  __shared__ float ss[Tt * Ll];

  int bw = blockIdx.x, tid = threadIdx.x;
  int b = bw >> 6;
  const float scale = 0.17677669529663689f;   // 32^-0.5

  for (int h = 0; h < NHh; ++h) {
    for (int idx = tid; idx < Tt * HDd; idx += 256) {
      int t = idx >> 5, d = idx & 31;
      sq[idx] = (float)q2[((size_t)bw * Tt + t) * Cc + h * HDd + d];
    }
    for (int idx = tid; idx < Ll * HDd; idx += 256) {
      int l = idx >> 5, d = idx & 31;
      size_t base = ((size_t)b * Ll + l) * Cc + h * HDd + d;
      sk[idx] = (float)k2[base];
      sv[idx] = (float)v2[base];
    }
    __syncthreads();

    for (int p = tid; p < Tt * Ll; p += 256) {
      int i = p >> 6, j = p & 63;
      float dot = 0.f;
      for (int d = 0; d < HDd; ++d) dot += sq[i * HDd + d] * sk[j * HDd + d];
      ss[p] = dot * scale;
    }
    __syncthreads();

    if (tid < Tt) {
      float m = -1e30f;
      for (int j = 0; j < Ll; ++j) m = fmaxf(m, ss[tid * Ll + j]);
      float s = 0.f;
      for (int j = 0; j < Ll; ++j) {
        float e = __expf(ss[tid * Ll + j] - m);
        ss[tid * Ll + j] = e; s += e;
      }
      float inv = 1.0f / s;
      for (int j = 0; j < Ll; ++j) ss[tid * Ll + j] *= inv;
    }
    __syncthreads();

    for (int idx = tid; idx < Tt * HDd; idx += 256) {
      int t = idx >> 5, d = idx & 31;
      float o = 0.f;
      for (int j = 0; j < Ll; ++j) o += ss[t * Ll + j] * sv[j * HDd + d];
      co[((size_t)bw * Tt + t) * Cc + h * HDd + d] = (__bf16)o;
    }
    __syncthreads();
  }
}

// ---------------------------------------------------------------------------
// Block-wide (512-elem) mean/var reduction helper.
// ---------------------------------------------------------------------------
DEVFN void block_reduce_2(float& s1, float& s2) {
  __shared__ float red[512];
  int tid = threadIdx.x;
  red[tid] = s1; red[tid + 256] = s2;
  __syncthreads();
  for (int off = 128; off > 0; off >>= 1) {
    if (tid < off) {
      red[tid] += red[tid + off];
      red[tid + 256] += red[tid + 256 + off];
    }
    __syncthreads();
  }
  s1 = red[0]; s2 = red[256];
  __syncthreads();
}

// skip = LN1(fold(roll(wproj_out))) + visual  — one block per token row.
__global__ __launch_bounds__(256) void k_merge_ln_skip(
    const float* __restrict__ wp, const float* __restrict__ visual,
    const float* __restrict__ g, const float* __restrict__ be,
    float* __restrict__ skip) {
  int rrow = blockIdx.x, tid = threadIdx.x;
  int b, h, w;
  token_to_pixel(rrow, b, h, w);
  float x0 = wp[(size_t)rrow * Cc + tid];
  float x1 = wp[(size_t)rrow * Cc + 256 + tid];
  float s1 = x0 + x1, s2 = x0 * x0 + x1 * x1;
  block_reduce_2(s1, s2);
  float mu = s1 * (1.0f / Cc);
  float rstd = rsqrtf(s2 * (1.0f / Cc) - mu * mu + 1e-5f);
  size_t base = ((size_t)b * Cc) * NPIX + (size_t)h * Wimg + w;
  int c0 = tid, c1 = tid + 256;
  skip[base + (size_t)c0 * NPIX] =
      (x0 - mu) * rstd * g[c0] + be[c0] + visual[base + (size_t)c0 * NPIX];
  skip[base + (size_t)c1 * NPIX] =
      (x1 - mu) * rstd * g[c1] + be[c1] + visual[base + (size_t)c1 * NPIX];
}

// skip2 = LN2(ca_out rows) folded+rolled + skip; also emits bf16 FFN operand
// in (b, h*56+w, c) row layout.
__global__ __launch_bounds__(256) void k_ln2_fold_add(
    const float* __restrict__ cout, const float* __restrict__ skip,
    const float* __restrict__ g, const float* __restrict__ be,
    float* __restrict__ skip2, __bf16* __restrict__ tff) {
  int rrow = blockIdx.x, tid = threadIdx.x;
  int b, h, w;
  token_to_pixel(rrow, b, h, w);
  float x0 = cout[(size_t)rrow * Cc + tid];
  float x1 = cout[(size_t)rrow * Cc + 256 + tid];
  float s1 = x0 + x1, s2 = x0 * x0 + x1 * x1;
  block_reduce_2(s1, s2);
  float mu = s1 * (1.0f / Cc);
  float rstd = rsqrtf(s2 * (1.0f / Cc) - mu * mu + 1e-5f);
  size_t base = ((size_t)b * Cc) * NPIX + (size_t)h * Wimg + w;
  size_t trow = ((size_t)b * NPIX + (size_t)h * Wimg + w) * Cc;
  int c0 = tid, c1 = tid + 256;
  float y0 = (x0 - mu) * rstd * g[c0] + be[c0] + skip[base + (size_t)c0 * NPIX];
  float y1 = (x1 - mu) * rstd * g[c1] + be[c1] + skip[base + (size_t)c1 * NPIX];
  skip2[base + (size_t)c0 * NPIX] = y0;
  skip2[base + (size_t)c1 * NPIX] = y1;
  tff[trow + c0] = (__bf16)y0;
  tff[trow + c1] = (__bf16)y1;
}

// out = skip2 + LN3(ffn rows) — rows are (b, h*56+w).
__global__ __launch_bounds__(256) void k_ln3_final(
    const float* __restrict__ ffo, const float* __restrict__ skip2,
    const float* __restrict__ g, const float* __restrict__ be,
    float* __restrict__ out) {
  int rrow = blockIdx.x, tid = threadIdx.x;
  int b = rrow / NPIX, p = rrow % NPIX;
  float x0 = ffo[(size_t)rrow * Cc + tid];
  float x1 = ffo[(size_t)rrow * Cc + 256 + tid];
  float s1 = x0 + x1, s2 = x0 * x0 + x1 * x1;
  block_reduce_2(s1, s2);
  float mu = s1 * (1.0f / Cc);
  float rstd = rsqrtf(s2 * (1.0f / Cc) - mu * mu + 1e-5f);
  size_t base = ((size_t)b * Cc) * NPIX + p;
  int c0 = tid, c1 = tid + 256;
  out[base + (size_t)c0 * NPIX] =
      (x0 - mu) * rstd * g[c0] + be[c0] + skip2[base + (size_t)c0 * NPIX];
  out[base + (size_t)c1 * NPIX] =
      (x1 - mu) * rstd * g[c1] + be[c1] + skip2[base + (size_t)c1 * NPIX];
}

// ---------------------------------------------------------------------------
extern "C" void kernel_launch(void* const* d_in, const int* in_sizes, int n_in,
                              void* d_out, int out_size, void* d_ws,
                              size_t ws_size, hipStream_t stream) {
  const float* visual   = (const float*)d_in[0];
  const float* vector_i = (const float*)d_in[1];
  const float* tm_w     = (const float*)d_in[2];
  const float* tm_b     = (const float*)d_in[3];
  const float* ln1_g    = (const float*)d_in[4];
  const float* ln1_b    = (const float*)d_in[5];
  const float* ln2_g    = (const float*)d_in[6];
  const float* ln2_b    = (const float*)d_in[7];
  const float* ln3_g    = (const float*)d_in[8];
  const float* ln3_b    = (const float*)d_in[9];
  const float* qkv_w    = (const float*)d_in[10];
  const float* qkv_b    = (const float*)d_in[11];
  const float* wproj_w  = (const float*)d_in[12];
  const float* wproj_b  = (const float*)d_in[13];
  const float* meta_w1  = (const float*)d_in[14];
  const float* meta_b1  = (const float*)d_in[15];
  const float* meta_w2  = (const float*)d_in[16];
  const float* meta_b2  = (const float*)d_in[17];
  const float* tau      = (const float*)d_in[18];
  const float* ca_in_w  = (const float*)d_in[19];
  const float* ca_in_b  = (const float*)d_in[20];
  const float* ca_out_w = (const float*)d_in[21];
  const float* ca_out_b = (const float*)d_in[22];
  const float* ff_w1    = (const float*)d_in[23];
  const float* ff_b1    = (const float*)d_in[24];
  const float* ff_w2    = (const float*)d_in[25];
  const float* ff_b2    = (const float*)d_in[26];

  // Workspace layout with lifetime-based reuse.
  char* base = (char*)d_ws;
  size_t off = 0;
  auto alloc = [&](size_t bytes) {
    size_t o = off;
    off += (bytes + 255) & ~(size_t)255;
    return base + o;
  };
  // R1: qkv(bf16) -> co(bf16) -> ffh(bf16)
  char* R1 = alloc((size_t)NTOK * 2048 * sizeof(__bf16));
  // R2: tokens -> tokens2 -> tff (all bf16 NTOKx512)
  char* R2 = alloc((size_t)NTOK * Cc * sizeof(__bf16));
  // R3: wout -> q2 (bf16 NTOKx512)
  char* R3 = alloc((size_t)NTOK * Cc * sizeof(__bf16));
  // R4: wp -> cout -> ffo (fp32 NTOKx512)
  char* R4 = alloc((size_t)NTOK * Cc * sizeof(float));
  // R5: skip (fp32 BCHW), R6: skip2 (fp32 BCHW)
  char* R5 = alloc((size_t)Bn * Cc * NPIX * sizeof(float));
  char* R6 = alloc((size_t)Bn * Cc * NPIX * sizeof(float));
  __bf16* vec_bf = (__bf16*)alloc((size_t)Bn * Ll * 256 * sizeof(__bf16));
  __bf16* vecg   = (__bf16*)alloc((size_t)Bn * Ll * Cc * sizeof(__bf16));
  __bf16* k2     = (__bf16*)alloc((size_t)Bn * Ll * Cc * sizeof(__bf16));
  __bf16* v2     = (__bf16*)alloc((size_t)Bn * Ll * Cc * sizeof(__bf16));
  float*  biasT  = (float*)alloc((size_t)NHh * Tt * Tt * sizeof(float));

  __bf16* tokens = (__bf16*)R2;
  __bf16* qkv    = (__bf16*)R1;
  __bf16* wout   = (__bf16*)R3;
  float*  wp     = (float*)R4;
  float*  skip   = (float*)R5;
  __bf16* q2     = (__bf16*)R3;
  __bf16* co     = (__bf16*)R1;
  float*  cout_  = (float*)R4;
  float*  skip2  = (float*)R6;
  __bf16* tff    = (__bf16*)R2;
  __bf16* ffh    = (__bf16*)R1;
  float*  ffo    = (float*)R4;

  const dim3 blk(256);

  // 1) vec = gelu(vector @ tm_w + tm_b) -> bf16   (M=512,K=256,N=512)
  k_f32_to_bf16<<<dim3((Bn * Ll * 256 + 255) / 256), blk, 0, stream>>>(
      vector_i, vec_bf, Bn * Ll * 256);
  k_gemm<1><<<dim3(Cc / BN, (Bn * Ll) / BM), blk, 0, stream>>>(
      vec_bf, tm_w, Cc, tm_b, vecg, Cc, 256);

  // 2) relative-position bias table (NH x 49 x 49)
  k_bias_mlp<<<dim3(Tt * Tt), blk, 0, stream>>>(meta_w1, meta_b1, meta_w2,
                                                meta_b2, biasT);

  // 3) tokens = roll+unfold(visual) -> bf16, then qkv GEMM (25088x512x1536)
  k_gather_tokens<<<dim3(NTOK), blk, 0, stream>>>(visual, tokens);
  k_gemm<0><<<dim3(3 * Cc / BN, NTOK / BM), blk, 0, stream>>>(
      tokens, qkv_w, 3 * Cc, qkv_b, qkv, 3 * Cc, Cc);

  // 4) window attention, then wproj GEMM (fp32 out)
  k_win_attn<<<dim3(BWw), blk, 0, stream>>>(qkv, biasT, tau, wout);
  k_gemm<2><<<dim3(Cc / BN, NTOK / BM), blk, 0, stream>>>(
      wout, wproj_w, Cc, wproj_b, wp, Cc, Cc);

  // 5) skip = LN1(fold+roll) + visual
  k_merge_ln_skip<<<dim3(NTOK), blk, 0, stream>>>(wp, visual, ln1_g, ln1_b,
                                                  skip);

  // 6) cross-attention projections
  k_gather_tokens<<<dim3(NTOK), blk, 0, stream>>>(skip, (__bf16*)R2);
  k_gemm<0><<<dim3(Cc / BN, NTOK / BM), blk, 0, stream>>>(
      (__bf16*)R2, ca_in_w, 3 * Cc, ca_in_b, q2, Cc, Cc);
  k_gemm<0><<<dim3(Cc / BN, (Bn * Ll) / BM), blk, 0, stream>>>(
      vecg, ca_in_w + Cc, 3 * Cc, ca_in_b + Cc, k2, Cc, Cc);
  k_gemm<0><<<dim3(Cc / BN, (Bn * Ll) / BM), blk, 0, stream>>>(
      vecg, ca_in_w + 2 * Cc, 3 * Cc, ca_in_b + 2 * Cc, v2, Cc, Cc);

  // 7) cross attention + output projection (fp32 out)
  k_cross_attn<<<dim3(BWw), blk, 0, stream>>>(q2, k2, v2, co);
  k_gemm<2><<<dim3(Cc / BN, NTOK / BM), blk, 0, stream>>>(
      co, ca_out_w, Cc, ca_out_b, cout_, Cc, Cc);

  // 8) skip2 = LN2 + fold/roll + skip; also emit FFN operand matrix
  k_ln2_fold_add<<<dim3(NTOK), blk, 0, stream>>>(cout_, skip, ln2_g, ln2_b,
                                                 skip2, tff);

  // 9) FFN: gelu(t @ ff_w1 + b1) @ ff_w2 + b2
  k_gemm<1><<<dim3(4 * Cc / BN, NTOK / BM), blk, 0, stream>>>(
      tff, ff_w1, 4 * Cc, ff_b1, ffh, 4 * Cc, Cc);
  k_gemm<2><<<dim3(Cc / BN, NTOK / BM), blk, 0, stream>>>(
      ffh, ff_w2, Cc, ff_b2, ffo, Cc, 4 * Cc);

  // 10) out = skip2 + LN3(ffn)
  k_ln3_final<<<dim3(NTOK), blk, 0, stream>>>(ffo, skip2, ln3_g, ln3_b,
                                              (float*)d_out);
}
